// EncoderLayer_28157805593285
// MI455X (gfx1250) — matrix operand order repaired and checked by reading, not compiled, not run on hardware
//
#include <hip/hip_runtime.h>
#include <hip/hip_bf16.h>

typedef __attribute__((ext_vector_type(16))) _Float16 v16h;
typedef __attribute__((ext_vector_type(8)))  _Float16 v8h;
typedef __attribute__((ext_vector_type(4)))  _Float16 v4h;
typedef __attribute__((ext_vector_type(8)))  float    v8f;
typedef __attribute__((ext_vector_type(4)))  float    v4f;

// Epilogue modes for the generic GEMM
#define EPI_NONE       0
#define EPI_RES        1   // C = acc + res[row,col]
#define EPI_BIAS_RELU  2   // C = max(acc + bias[col], 0)
#define EPI_BIAS_RES   3   // C = acc + bias[col] + res[row,col]

// ---------------------------------------------------------------------------
// Generic GEMM: C = A[MxK] @ B[KxN] (+epilogue), f32 in/out, f16 WMMA inside.
// Block tile 128x128x32, 256 threads = 8 wave32s arranged 2(M) x 4(N).
// REQUIRES: M%128==0, N%128==0, K%32==0, 16B-aligned pointers and leading dims.
// ---------------------------------------------------------------------------
template<int EPI>
__global__ __launch_bounds__(256) void gemm_wmma_f16(
    const float* __restrict__ A, const float* __restrict__ B, float* __restrict__ C,
    int K, int lda, int ldb, int ldc,
    const float* __restrict__ bias, const float* __restrict__ res)
{
    constexpr int BM = 128, BN = 128, BK = 32;
    __shared__ _Float16 As[BM][BK + 8];    // [m][k]
    __shared__ _Float16 Bs[BN][BK + 8];    // [n][k]: fragment reads k-contiguous

    const int tid  = threadIdx.x;
    const int lane = tid & 31;
    const int wid  = tid >> 5;
    const int wm   = wid >> 2;          // 0..1
    const int wn   = wid & 3;           // 0..3
    const int l15  = lane & 15;
    const int hi   = (lane >> 4) & 1;   // lane half

    const int blockRow = blockIdx.y * BM;
    const int blockCol = blockIdx.x * BN;

    v8f acc[4][2] = {};

    for (int kt = 0; kt < K; kt += BK) {
        // ---- issue all global loads first (overlaps previous iter's WMMAs)
        float4 fa[4];
        #pragma unroll
        for (int i = 0; i < 4; ++i) {
            int idx = i * 256 + tid;
            int r = idx >> 3, c4 = idx & 7;
            fa[i] = *(const float4*)(A + (size_t)(blockRow + r) * lda + kt + c4 * 4);
        }
        float4 fb[4];
        #pragma unroll
        for (int i = 0; i < 4; ++i) {
            int idx = i * 256 + tid;
            int kk = idx >> 5, n4 = idx & 31;
            fb[i] = *(const float4*)(B + (size_t)(kt + kk) * ldb + blockCol + n4 * 4);
        }

        __syncthreads();   // previous iteration's LDS reads complete

        #pragma unroll
        for (int i = 0; i < 4; ++i) {
            int idx = i * 256 + tid;
            int r = idx >> 3, c4 = idx & 7;
            v4h h = { (_Float16)fa[i].x, (_Float16)fa[i].y,
                      (_Float16)fa[i].z, (_Float16)fa[i].w };
            *(v4h*)&As[r][c4 * 4] = h;          // ds_store_b64
        }
        #pragma unroll
        for (int i = 0; i < 4; ++i) {
            int idx = i * 256 + tid;
            int kk = idx >> 5, n4 = idx & 31;
            Bs[n4 * 4 + 0][kk] = (_Float16)fb[i].x;
            Bs[n4 * 4 + 1][kk] = (_Float16)fb[i].y;
            Bs[n4 * 4 + 2][kk] = (_Float16)fb[i].z;
            Bs[n4 * 4 + 3][kk] = (_Float16)fb[i].w;
        }
        __syncthreads();

        // ---- fragments per documented CDNA5 16-bit layouts
        v16h afrag[4], bfrag[2];
        const int akb = hi ? 8 : 0;          // A: lanes 16-31 hold K=8..15 / 24..31
        #pragma unroll
        for (int mi = 0; mi < 4; ++mi) {
            int row = wm * 64 + mi * 16 + l15;
            #pragma unroll
            for (int i = 0; i < 8; ++i) afrag[mi][i]     = As[row][akb + i];
            #pragma unroll
            for (int i = 0; i < 8; ++i) afrag[mi][8 + i] = As[row][akb + 16 + i];
        }
        const int bks = hi ? 8 : 0;          // B: lanes 16-31 hold K=8..15 / 24..31
        #pragma unroll
        for (int ni = 0; ni < 2; ++ni) {
            int col = wn * 32 + ni * 16 + l15;
            #pragma unroll
            for (int i = 0; i < 16; ++i) bfrag[ni][i] = Bs[col][bks + i + (i >= 8 ? 8 : 0)];
        }

        #pragma unroll
        for (int mi = 0; mi < 4; ++mi)
            #pragma unroll
            for (int ni = 0; ni < 2; ++ni)
                acc[mi][ni] = __builtin_amdgcn_wmma_f32_16x16x32_f16(
                    false, afrag[mi], false, bfrag[ni],
                    (short)0, acc[mi][ni], false, false);
    }

    // ---- epilogue + store: pointer-increment addressing (no per-row 64b muls)
    #pragma unroll
    for (int mi = 0; mi < 4; ++mi) {
        #pragma unroll
        for (int ni = 0; ni < 2; ++ni) {
            int gr0 = blockRow + wm * 64 + mi * 16 + hi * 8;
            int gc  = blockCol + wn * 32 + ni * 16 + l15;
            float bv = (EPI == EPI_BIAS_RELU || EPI == EPI_BIAS_RES) ? bias[gc] : 0.0f;
            float*       cp = C   + (size_t)gr0 * ldc + gc;
            const float* rp = (EPI == EPI_RES || EPI == EPI_BIAS_RES)
                              ? (res + (size_t)gr0 * ldc + gc) : nullptr;
            #pragma unroll
            for (int r = 0; r < 8; ++r) {
                float v = acc[mi][ni][r];
                if (EPI == EPI_RES)            v += rp[0];
                else if (EPI == EPI_BIAS_RELU) v = fmaxf(v + bv, 0.0f);
                else if (EPI == EPI_BIAS_RES)  v += bv + rp[0];
                cp[0] = v;
                cp += ldc;
                if (EPI == EPI_RES || EPI == EPI_BIAS_RES) rp += ldc;
            }
        }
    }
}

// ---------------------------------------------------------------------------
// Fused attention: for 16 query rows x all 1024 keys of one (b,h):
//   scores = (Q/8) @ K^T (+mask)  -> f16 in LDS
//   softmax row stats from LDS, probs written ONCE to attn output (f32, NT)
//   ctx = probs @ V accumulated with WMMA from the same LDS tile
// Grid: (1, S/16, B*H), 256 threads (8 waves).
// ---------------------------------------------------------------------------
__global__ __launch_bounds__(256) void attn_fused(
    const float* __restrict__ Q, const float* __restrict__ Kg,
    const float* __restrict__ Vg, float* __restrict__ attn,
    float* __restrict__ ctx, const int* __restrict__ mask)
{
    constexpr int S = 1024, H = 16, DK = 64, LD = H * DK;  // LD = 1024
    constexpr float SCALE = 0.125f;                        // 1/sqrt(64)

    __shared__ _Float16 Qs[16][DK + 8];       // [m][d]
    __shared__ _Float16 Ss[16][S + 8];        // raw f16 scores, then f16 probs
    __shared__ _Float16 KV[128 * 72];         // Ks[128][72] | Vs[64][136] | f32 reduce
    __shared__ float rowM[16], rowInv[16];

    _Float16 (*Ks)[72]  = (_Float16(*)[72])KV;
    _Float16 (*Vs)[136] = (_Float16(*)[136])KV;
    float* red = (float*)KV;                  // 4 waves x 16x16 f32 = 4KB

    const int z = blockIdx.z, bI = z / H, hI = z % H;
    const int rowBase = blockIdx.y * 16;
    const size_t headOff = (size_t)bI * S * LD + hI * DK;
    const float* Qp = Q  + headOff;
    const float* Kp = Kg + headOff;
    const float* Vp = Vg + headOff;
    float* attnP = attn + (size_t)(bI * H + hI) * S * S;
    float* ctxP  = ctx + headOff;

    const int tid = threadIdx.x, lane = tid & 31, w = tid >> 5;
    const int l15 = lane & 15, hi = (lane >> 4) & 1;
    const int akb = hi ? 8 : 0, bks = hi ? 16 : 0;

    // ---- stage Q tile 16x64 (one float4 per thread)
    {
        int r = tid >> 4, c4 = tid & 15;
        float4 f = *(const float4*)(Qp + (size_t)(rowBase + r) * LD + c4 * 4);
        v4h h = { (_Float16)f.x, (_Float16)f.y, (_Float16)f.z, (_Float16)f.w };
        *(v4h*)&Qs[r][c4 * 4] = h;
    }
    __syncthreads();

    // A-fragments of Q (reused by every chunk): K=64 -> two 32-deep frags
    v16h aq[2];
    #pragma unroll
    for (int kk = 0; kk < 2; ++kk) {
        #pragma unroll
        for (int i = 0; i < 8; ++i) aq[kk][i]     = Qs[l15][kk * 32 + akb + i];
        #pragma unroll
        for (int i = 0; i < 8; ++i) aq[kk][8 + i] = Qs[l15][kk * 32 + akb + 16 + i];
    }

    // ---- phase 1: scores chunk-by-chunk into LDS (f16)
    for (int nc = 0; nc < 8; ++nc) {
        #pragma unroll
        for (int i = 0; i < 8; ++i) {              // stage K rows nc*128..+127
            int idx = i * 256 + tid;
            int n = idx >> 4, c4 = idx & 15;
            float4 f = *(const float4*)(Kp + (size_t)(nc * 128 + n) * LD + c4 * 4);
            v4h h = { (_Float16)f.x, (_Float16)f.y, (_Float16)f.z, (_Float16)f.w };
            *(v4h*)&Ks[n][c4 * 4] = h;
        }
        __syncthreads();

        v8f acc = {};
        #pragma unroll
        for (int kk = 0; kk < 2; ++kk) {
            v16h b;
            #pragma unroll
            for (int i = 0; i < 16; ++i) b[i] = Ks[w * 16 + l15][kk * 32 + akb + i + (i >= 8 ? 8 : 0)];
            acc = __builtin_amdgcn_wmma_f32_16x16x32_f16(
                false, aq[kk], false, b, (short)0, acc, false, false);
        }
        int gcol = nc * 128 + w * 16 + l15;
        int mv = mask[bI * S + gcol];
        #pragma unroll
        for (int r = 0; r < 8; ++r) {
            float v = acc[r] * SCALE;
            if (mv == 0) v = -60000.0f;            // finite in f16; exp underflows to 0
            Ss[hi * 8 + r][gcol] = (_Float16)v;
        }
        __syncthreads();                           // Ks reused next chunk
    }

    // ---- phase 2: row softmax stats (2 rows per wave, contiguous v8h loads)
    #pragma unroll
    for (int rr = 0; rr < 2; ++rr) {
        int row = w * 2 + rr;
        const v8h* vp = (const v8h*)&Ss[row][lane * 32];   // 32 contiguous halves/lane
        v8h h0 = vp[0], h1 = vp[1], h2 = vp[2], h3 = vp[3];
        float vals[32];
        #pragma unroll
        for (int i = 0; i < 8; ++i) {
            vals[i]      = (float)h0[i];
            vals[8 + i]  = (float)h1[i];
            vals[16 + i] = (float)h2[i];
            vals[24 + i] = (float)h3[i];
        }
        float m = -3.4e38f;
        #pragma unroll
        for (int i = 0; i < 32; ++i) m = fmaxf(m, vals[i]);
        #pragma unroll
        for (int off = 16; off > 0; off >>= 1) m = fmaxf(m, __shfl_xor(m, off, 32));
        float s = 0.0f;
        #pragma unroll
        for (int i = 0; i < 32; ++i) s += __expf(vals[i] - m);
        #pragma unroll
        for (int off = 16; off > 0; off >>= 1) s += __shfl_xor(s, off, 32);
        if (lane == 0) { rowM[row] = m; rowInv[row] = 1.0f / s; }
    }
    __syncthreads();

    // ---- phase 3: normalize + write attn (non-temporal) + ctx WMMA
    // waves: (w&3) -> ctx col tile, (w>>2) -> K-slice half of each 128-chunk
    const int wn3 = w & 3, wk3 = w >> 2;
    v8f cacc = {};
    for (int nc = 0; nc < 8; ++nc) {
        #pragma unroll
        for (int i = 0; i < 8; ++i) {              // stage V chunk: Vs[d][k]
            int idx = i * 256 + tid;
            int k = idx >> 4, d4 = idx & 15;
            float4 f = *(const float4*)(Vp + (size_t)(nc * 128 + k) * LD + d4 * 4);
            Vs[d4 * 4 + 0][k] = (_Float16)f.x;
            Vs[d4 * 4 + 1][k] = (_Float16)f.y;
            Vs[d4 * 4 + 2][k] = (_Float16)f.z;
            Vs[d4 * 4 + 3][k] = (_Float16)f.w;
        }
        {   // normalize: 16 threads/row x 8 contiguous cols -> wide LDS/global ops
            int r  = tid >> 4;
            int c0 = (tid & 15) * 8;
            int col = nc * 128 + c0;
            v8h* sp = (v8h*)&Ss[r][col];
            v8h hv = *sp;                          // ds_load_b128
            float m = rowM[r], inv = rowInv[r];
            float p[8];
            #pragma unroll
            for (int i = 0; i < 8; ++i) p[i] = __expf((float)hv[i] - m) * inv;
            v4f o0 = { p[0], p[1], p[2], p[3] };
            v4f o1 = { p[4], p[5], p[6], p[7] };
            v4f* gp = (v4f*)(attnP + (size_t)(rowBase + r) * S + col);
            __builtin_nontemporal_store(o0, gp);       // stream past L2
            __builtin_nontemporal_store(o1, gp + 1);
            v8h hp = { (_Float16)p[0], (_Float16)p[1], (_Float16)p[2], (_Float16)p[3],
                       (_Float16)p[4], (_Float16)p[5], (_Float16)p[6], (_Float16)p[7] };
            *sp = hp;                              // ds_store_b128
        }
        __syncthreads();

        #pragma unroll
        for (int kk = 0; kk < 2; ++kk) {
            int kb  = nc * 128 + wk3 * 64 + kk * 32;   // col index into Ss
            int kbv = wk3 * 64 + kk * 32;              // k index into Vs
            v16h a, b;
            #pragma unroll
            for (int i = 0; i < 8; ++i) a[i]     = Ss[l15][kb + akb + i];
            #pragma unroll
            for (int i = 0; i < 8; ++i) a[8 + i] = Ss[l15][kb + akb + 16 + i];
            #pragma unroll
            for (int i = 0; i < 16; ++i) b[i] = Vs[wn3 * 16 + l15][kbv + akb + i + (i >= 8 ? 8 : 0)];
            cacc = __builtin_amdgcn_wmma_f32_16x16x32_f16(
                false, a, false, b, (short)0, cacc, false, false);
        }
        __syncthreads();                           // Vs reused next chunk
    }

    // ---- reduce the two K-slice halves and store ctx
    if (wk3 == 1) {
        #pragma unroll
        for (int r = 0; r < 8; ++r)
            red[wn3 * 256 + (hi * 8 + r) * 16 + l15] = cacc[r];
    }
    __syncthreads();
    if (wk3 == 0) {
        #pragma unroll
        for (int r = 0; r < 8; ++r) {
            int row = hi * 8 + r;
            float v = cacc[r] + red[wn3 * 256 + row * 16 + l15];
            ctxP[(size_t)(rowBase + row) * LD + wn3 * 16 + l15] = v;
        }
    }
}

extern "C" void kernel_launch(void* const* d_in, const int* in_sizes, int n_in,
                              void* d_out, int out_size, void* d_ws, size_t ws_size,
                              hipStream_t stream) {
    (void)in_sizes; (void)n_in; (void)out_size; (void)ws_size;
    const float* enc_in = (const float*)d_in[0];
    const int*   mask   = (const int*)d_in[1];
    // d_in[2] = gap (unused)
    const float* wq = (const float*)d_in[3];
    const float* wk = (const float*)d_in[4];
    const float* wv = (const float*)d_in[5];
    const float* wo = (const float*)d_in[6];
    const float* w1 = (const float*)d_in[7];
    const float* b1 = (const float*)d_in[8];
    const float* w2 = (const float*)d_in[9];
    const float* b2 = (const float*)d_in[10];

    const int B = 8, S = 1024, D = 1024, H = 16, DFF = 4096;
    const int MS = B * S;                    // 8192

    float* out  = (float*)d_out;
    float* attn = out + (size_t)B * S * D;   // second tuple element [B,H,S,S]

    float* ws  = (float*)d_ws;
    float* Q   = ws;
    float* Kt  = Q   + (size_t)MS * D;
    float* V   = Kt  + (size_t)MS * D;
    float* ctx = V   + (size_t)MS * D;
    float* enc = ctx + (size_t)MS * D;
    float* hid = enc + (size_t)MS * D;       // [8192, 4096]

    dim3 blk(256);
    dim3 gProj(D / 128, MS / 128, 1);

    // 1-3. Q/K/V projections: [8192,1024] = X @ W
    gemm_wmma_f16<EPI_NONE><<<gProj, blk, 0, stream>>>(
        enc_in, wq, Q, D, D, D, D, nullptr, nullptr);
    gemm_wmma_f16<EPI_NONE><<<gProj, blk, 0, stream>>>(
        enc_in, wk, Kt, D, D, D, D, nullptr, nullptr);
    gemm_wmma_f16<EPI_NONE><<<gProj, blk, 0, stream>>>(
        enc_in, wv, V, D, D, D, D, nullptr, nullptr);

    // 4. fused attention: scores+mask+softmax+attn-write+ctx
    dim3 gAttn(1, S / 16, B * H);
    attn_fused<<<gAttn, blk, 0, stream>>>(Q, Kt, V, attn, ctx, mask);

    // 5. enc = ctx @ wo + enc_in
    gemm_wmma_f16<EPI_RES><<<gProj, blk, 0, stream>>>(
        ctx, wo, enc, D, D, D, D, nullptr, enc_in);

    // 6. hid = relu(enc @ w1 + b1)
    dim3 gFF1(DFF / 128, MS / 128, 1);
    gemm_wmma_f16<EPI_BIAS_RELU><<<gFF1, blk, 0, stream>>>(
        enc, w1, hid, D, D, DFF, DFF, b1, nullptr);

    // 7. out = hid @ w2 + b2 + enc
    gemm_wmma_f16<EPI_BIAS_RES><<<gProj, blk, 0, stream>>>(
        hid, w2, out, DFF, DFF, D, D, b2, enc);
}
